// NeuralCDE_55628416417903
// MI455X (gfx1250) — matrix operand order repaired and checked by reading, not compiled, not run on hardware
//
#include <hip/hip_runtime.h>
#include <hip/hip_bf16.h>

typedef __attribute__((ext_vector_type(16))) _Float16 v16h;
typedef __attribute__((ext_vector_type(8)))  _Float16 v8h;
typedef __attribute__((ext_vector_type(8)))  float    v8f;

#define LM1      127
#define HID      64
#define MLPH     128
#define CIN      8
#define TILE_M   16
#define NTHREADS 256

// ---------------- CDNA5 TDM availability probing ----------------------------
#if defined(__has_builtin)
#if __has_builtin(__builtin_amdgcn_tensor_load_to_lds) && \
    __has_builtin(__builtin_amdgcn_s_wait_tensorcnt)
#define CDE_TDM 1
#if __has_include(<hip/amd_detail/amd_gfx1250_TDM.h>)
#define CDE_TDM_6ARG 1            // therock-10.0 headers -> clang-23 6-arg form
#endif
#endif
#endif

#if defined(CDE_TDM)
typedef unsigned int u32x4 __attribute__((ext_vector_type(4)));
typedef int          i32x4 __attribute__((ext_vector_type(4)));
typedef int          i32x8 __attribute__((ext_vector_type(8)));

// Issue a TDM DMA: copy `ndw` dwords from global `gsrc` into LDS offset
// `lds_off`. Described as a 2D tensor with a single row of ndw elements
// (data_size=4B), per CDNA5 ISA D# group layout.
__device__ __forceinline__ void tdm_copy_to_lds(const void* gsrc,
                                                unsigned lds_off,
                                                unsigned ndw) {
  unsigned long long ga = (unsigned long long)gsrc;
  u32x4 g0;
  g0[0] = 1u;                                           // count=1 (valid), user mode
  g0[1] = lds_off;                                      // lds_addr (bytes)
  g0[2] = (unsigned)(ga & 0xFFFFFFFFu);                 // global_addr[31:0]
  g0[3] = (unsigned)((ga >> 32) & 0x01FFFFFFu) | (2u << 30); // addr[56:32], type=2
  i32x8 g1;
  g1[0] = (int)(2u << 16);                              // wg_mask=0, data_size=4B
  g1[1] = (int)((ndw & 0xFFFFu) << 16);                 // tensor_dim0[15:0]
  g1[2] = (int)((ndw >> 16) & 0xFFFFu) | (1 << 16);     // tensor_dim0[31:16], tensor_dim1=1
  g1[3] = (int)(ndw << 16);                             // tile_dim0 (<=65535 dwords)
  g1[4] = 1;                                            // tile_dim1=1, tile_dim2=0
  g1[5] = (int)ndw;                                     // tensor_dim0_stride[31:0]
  g1[6] = (int)((ndw & 0xFFFFu) << 16);                 // stride hi=0 | dim1_stride[15:0]
  g1[7] = (int)(ndw >> 16);                             // dim1_stride[47:16]
  i32x4 z4 = {};
#if defined(CDE_TDM_6ARG)
  i32x8 z8 = {};
  __builtin_amdgcn_tensor_load_to_lds(g0, g1, z4, z4, z8, 0);
#else
  __builtin_amdgcn_tensor_load_to_lds(g0, g1, z4, z4, 0);
#endif
}
#endif

// ---------------- fast tanh: 1 - 2/(exp2(2*log2e*x)+1) ----------------------
__device__ __forceinline__ float fast_tanh(float x) {
#if defined(__has_builtin) && __has_builtin(__builtin_amdgcn_exp2f) && \
    __has_builtin(__builtin_amdgcn_rcpf)
  float e = __builtin_amdgcn_exp2f(x * 2.885390081777927f);   // 2*log2(e)
  return 1.0f - 2.0f * __builtin_amdgcn_rcpf(e + 1.0f);
#else
  return tanhf(x);
#endif
}

__device__ __forceinline__ v16h cat8(v8h lo, v8h hi) {
  v16h r;
#pragma unroll
  for (int e = 0; e < 8; ++e) { r[e] = lo[e]; r[e + 8] = hi[e]; }
  return r;
}

__device__ __forceinline__ v8f wmma16(v16h a, v16h b, v8f c) {
  // D = A(16x32 f16) x B(32x16 f16) + C(16x16 f32)
  return __builtin_amdgcn_wmma_f32_16x16x32_f16(false, a, false, b, (short)0, c,
                                                false, false);
}

// ---------------- weight packing (f32 -> f16, WMMA B-fragment lane order) ----
// B fragment layout (wave32, 16-bit, 32x16): element e of lane l holds
//   K = kc*32 + ((l>=16)?8:0) + ((e<8)?e:e+8) ,  N = ntile*16 + (l&15)

__global__ void pack_w1(const float* __restrict__ W1, _Float16* __restrict__ out) {
  int id = blockIdx.x * 256 + threadIdx.x;      // 8*2*32*16 = 8192
  int e = id & 15, lane = (id >> 4) & 31, t = id >> 9;   // t = nt*2 + kc
  int kc = t & 1, nt = t >> 1;
  int k = kc * 32 + ((lane >= 16) ? 8 : 0) + ((e < 8) ? e : e + 8);  // 0..63  (row 1..64 of W1)
  int n = nt * 16 + (lane & 15);                         // 0..127
  out[id] = (_Float16)W1[(1 + k) * MLPH + n];
}

__global__ void pack_w2(const float* __restrict__ W2, _Float16* __restrict__ out) {
  int id = blockIdx.x * 256 + threadIdx.x;      // 32*4*32*16 = 65536
  int e = id & 15, lane = (id >> 4) & 31, t = id >> 9;   // t = nt*4 + kc
  int kc = t & 3, nt = t >> 2;
  int k = kc * 32 + ((lane >= 16) ? 8 : 0) + ((e < 8) ? e : e + 8);  // 0..127
  int n = nt * 16 + (lane & 15);                         // 0..511
  out[id] = (_Float16)W2[k * 512 + n];
}

// ---------------- main CDE integrator: 16 batch rows / workgroup ------------
__global__ __launch_bounds__(NTHREADS, 1) void cde_main(
    const float* __restrict__ coeffs,
    const float* __restrict__ Wi1, const float* __restrict__ bi1,
    const float* __restrict__ Wi2, const float* __restrict__ bi2,
    const float* __restrict__ W1,  const float* __restrict__ b1,
    const float* __restrict__ b2,
    const _Float16* __restrict__ w1p, const _Float16* __restrict__ w2p,
    float* __restrict__ out)
{
  __shared__ __align__(32) _Float16 sW1[8 * 2 * 32 * 16];    // 16 KB
  __shared__ __align__(32) _Float16 sW2[32 * 4 * 32 * 16];   // 128 KB
  __shared__ _Float16 sB1h[128];
  __shared__ _Float16 sW1th[128];
  __shared__ _Float16 sB2h[512];
  __shared__ __align__(16) float sZ[16 * 64];                // z state (f32)
  __shared__ __align__(16) float sK[16 * 64];                // current k stage
  __shared__ __align__(32) _Float16 sH[16 * 128];            // relu hidden (f16)
  __shared__ float sdX[3 * 16 * 8];                          // dX at fr=0,0.5,(k4)

  const int tid  = threadIdx.x;
  const int lane = tid & 31;
  const int wid  = tid >> 5;
  const int b0   = blockIdx.x * TILE_M;

  const int M    = lane & 15;          // A-frag row / C-frag column role
  const int loff = (lane >> 4) * 8;    // A-frag K offset for upper half-wave
  const int hiM  = (lane >> 4) * 8;    // C-frag row offset for upper half-wave

  // ---- one-time: packed weights -> LDS (TDM DMA if available) ----
#if defined(CDE_TDM)
  if (wid == 0) {
    tdm_copy_to_lds(w2p, (unsigned)(unsigned long long)(const void*)sW2, 32768u);
    tdm_copy_to_lds(w1p, (unsigned)(unsigned long long)(const void*)sW1, 4096u);
    __builtin_amdgcn_s_wait_tensorcnt(0);   // LDS writes done before barrier
  }
#else
  {
    const uint4* s2 = (const uint4*)w2p;  uint4* d2 = (uint4*)sW2;
    for (int i = tid; i < 8192; i += NTHREADS) d2[i] = s2[i];
    const uint4* s1 = (const uint4*)w1p;  uint4* d1 = (uint4*)sW1;
    for (int i = tid; i < 1024; i += NTHREADS) d1[i] = s1[i];
  }
#endif
  for (int i = tid; i < 128; i += NTHREADS) { sB1h[i] = (_Float16)b1[i]; sW1th[i] = (_Float16)W1[i]; }
  for (int i = tid; i < 512; i += NTHREADS) sB2h[i] = (_Float16)b2[i];

  // ---- z0 = tanh(relu(X0@Wi1+bi1)@Wi2+bi2), X0 = coeffs[:,0,0:8] ----
  float* sH20 = sdX;                   // reuse (needs 320 floats <= 384)
  for (int i = tid; i < 16 * 20; i += NTHREADS) {
    int r = i / 20, j = i % 20;
    const float* x0 = &coeffs[(size_t)(b0 + r) * LM1 * 32];
    float s = bi1[j];
#pragma unroll
    for (int c = 0; c < CIN; ++c) s += x0[c] * Wi1[c * 20 + j];
    sH20[i] = fmaxf(s, 0.f);
  }
  __syncthreads();
#pragma unroll
  for (int q = 0; q < 4; ++q) {
    int e = tid * 4 + q, r = e >> 6, hh = e & 63;
    float s = bi2[hh];
    for (int j = 0; j < 20; ++j) s += sH20[r * 20 + j] * Wi2[j * 64 + hh];
    float z = fast_tanh(s);
    sZ[e] = z;
    out[(size_t)(b0 + r) * (128 * 64) + hh] = z;
  }
  __syncthreads();

  // ---- 127 RK4 steps ----
  for (int i = 0; i < LM1; ++i) {
    // dX/dt at fr=0 (k1), fr=0.5 (k2,k3), and at t=i+1 (k4)
    if (tid < 128) {
      int r = tid >> 3, c = tid & 7;
      size_t base = ((size_t)(b0 + r) * LM1 + i) * 32;
      float bv = coeffs[base + 8 + c], cv = coeffs[base + 16 + c], dv = coeffs[base + 24 + c];
      sdX[0 * 128 + tid] = bv;
      sdX[1 * 128 + tid] = bv + cv + 0.75f * dv;
      if (i < LM1 - 1) {
        sdX[2 * 128 + tid] = coeffs[base + 32 + 8 + c];     // b[:, i+1], fr=0
        __builtin_prefetch(&coeffs[base + 32], 0, 1);       // next-step coeffs
      } else {
        sdX[2 * 128 + tid] = bv + 2.f * cv + 3.f * dv;      // fr=1 at last interval
      }
    }

    float acc[4] = {0.f, 0.f, 0.f, 0.f};

#pragma unroll
    for (int s = 0; s < 4; ++s) {
      const float alpha = (s == 0) ? 0.f : ((s == 3) ? 1.f : 0.5f);
      const float ts    = (float)i + ((s == 0) ? 0.f : ((s == 3) ? 1.f : 0.5f));
      const int   sel   = (s == 0) ? 0 : ((s == 3) ? 2 : 1);
      const float wgt   = (s == 1 || s == 2) ? 2.f : 1.f;

      // ---- GEMM1: h = relu((z + alpha*k) @ W1z + (b1 + t*W1t)), tile nt=wid ----
      {
        v8f c = {};
#pragma unroll
        for (int kc = 0; kc < 2; ++kc) {
          // A fragment built on the fly from f32 z (+ alpha*k)
          int base0 = M * 64 + kc * 32 + loff;      // 8 contiguous, then +16
          v16h a;
#pragma unroll
          for (int e = 0; e < 16; ++e) {
            int idx = base0 + ((e < 8) ? e : (8 + e));   // e>=8 -> +16 skip
            float z = sZ[idx];
            if (s != 0) z += alpha * sK[idx];
            a[e] = (_Float16)z;
          }
          v16h bf = *(const v16h*)&sW1[((wid * 2 + kc) * 32 + lane) * 16];
          c = wmma16(a, bf, c);
        }
        int n = wid * 16 + M;
        float bias = (float)sB1h[n] + ts * (float)sW1th[n];
#pragma unroll
        for (int r = 0; r < 8; ++r) {
          float h = fmaxf(c[r] + bias, 0.f);
          sH[(r + hiM) * 128 + n] = (_Float16)h;
        }
      }
      __syncthreads();

      // ---- GEMM2 + tanh + einsum('bhc,bc->bh'): 4 N-tiles per wave ----
      {
        v16h afr[4];
#pragma unroll
        for (int kc = 0; kc < 4; ++kc) {
          v8h lo = *(const v8h*)&sH[M * 128 + kc * 32 + loff];
          v8h hi = *(const v8h*)&sH[M * 128 + kc * 32 + loff + 16];
          afr[kc] = cat8(lo, hi);
        }
#pragma unroll
        for (int jj = 0; jj < 4; ++jj) {
          int nt = wid * 4 + jj;
          v8f c = {};
#pragma unroll
          for (int kc = 0; kc < 4; ++kc) {
            v16h bf = *(const v16h*)&sW2[((nt * 4 + kc) * 32 + lane) * 16];
            c = wmma16(afr[kc], bf, c);
          }
          int n = nt * 16 + M;            // global column, h = n>>3, c = n&7
#pragma unroll
          for (int r = 0; r < 8; ++r) {
            int Mr = r + hiM;
            float v = fast_tanh(c[r] + (float)sB2h[n]);
            float p = v * sdX[sel * 128 + Mr * 8 + (n & 7)];
            p += __shfl_xor(p, 1);
            p += __shfl_xor(p, 2);
            p += __shfl_xor(p, 4);        // sum over 8 channels (8-lane group)
            if ((lane & 7) == 0) sK[Mr * 64 + (n >> 3)] = p;
          }
        }
      }
      __syncthreads();

#pragma unroll
      for (int q = 0; q < 4; ++q) acc[q] += wgt * sK[tid * 4 + q];
    }

    // z_{i+1} = z + (k1 + 2k2 + 2k3 + k4)/6 ; emit to output
#pragma unroll
    for (int q = 0; q < 4; ++q) {
      int e = tid * 4 + q, r = e >> 6, hh = e & 63;
      float zn = sZ[e] + acc[q] * (1.f / 6.f);
      sZ[e] = zn;
      out[(size_t)(b0 + r) * (128 * 64) + (size_t)(i + 1) * 64 + hh] = zn;
    }
    __syncthreads();
  }
}

extern "C" void kernel_launch(void* const* d_in, const int* in_sizes, int n_in,
                              void* d_out, int out_size, void* d_ws, size_t ws_size,
                              hipStream_t stream) {
  (void)in_sizes; (void)n_in; (void)out_size; (void)ws_size;
  const float* coeffs = (const float*)d_in[0];
  const float* Wi1    = (const float*)d_in[1];
  const float* bi1    = (const float*)d_in[2];
  const float* Wi2    = (const float*)d_in[3];
  const float* bi2    = (const float*)d_in[4];
  const float* W1     = (const float*)d_in[5];
  const float* b1     = (const float*)d_in[6];
  const float* W2     = (const float*)d_in[7];
  const float* b2     = (const float*)d_in[8];

  _Float16* w1p = (_Float16*)d_ws;                                     // 16384 B
  _Float16* w2p = (_Float16*)((char*)d_ws + 8 * 2 * 32 * 16 * 2);      // 131072 B

  pack_w1<<<32,  256, 0, stream>>>(W1, w1p);
  pack_w2<<<256, 256, 0, stream>>>(W2, w2p);
  cde_main<<<64, NTHREADS, 0, stream>>>(coeffs, Wi1, bi1, Wi2, bi2,
                                        W1, b1, b2, w1p, w2p, (float*)d_out);
}
